// HierarchicalCNN_53575422050797
// MI455X (gfx1250) — matrix operand order repaired and checked by reading, not compiled, not run on hardware
//
#include <hip/hip_runtime.h>
#include <hip/hip_bf16.h>

typedef __attribute__((ext_vector_type(16))) __bf16 v16bf;
typedef __attribute__((ext_vector_type(8)))  float  v8f;

union FragB { v16bf v; __bf16 e[16]; uint4 q[2]; };
union Pack8 { uint4 q; __bf16 e[8]; };

// A-matrix (16x32 bf16) element->K mapping used by this file (contiguous, same as B loads):
// element e (0..15), lane: K = (hi<<4) | e, M = lane&15
__device__ __host__ inline int a_kidx(int lane, int e) {
    return (((lane >> 4) & 1) << 4) | e;
}

__device__ inline v8f wmma_bf16(v16bf a, v16bf b, v8f c) {
    return __builtin_amdgcn_wmma_f32_16x16x32_bf16(
        /*neg_a=*/false, a, /*neg_b=*/false, b,
        /*c_mod=*/(short)0, c, /*reuse_a=*/false, /*reuse_b=*/false);
}

// ---------------- weight prep: swizzle fp32 conv weights into bf16 A-fragments ----
// per-CNN fragment elems: conv0 4*512, conv1 80*512, conv2 320*512 = 206848 total
#define PER_CNN 206848

__global__ void prep_weights(const float* __restrict__ w0a, const float* __restrict__ w1a,
                             const float* __restrict__ w2a,
                             const float* __restrict__ w0b, const float* __restrict__ w1b,
                             const float* __restrict__ w2b,
                             __bf16* __restrict__ wfrag) {
    int gid = blockIdx.x * blockDim.x + threadIdx.x;
    if (gid >= 2 * PER_CNN) return;
    int cnn = gid / PER_CNN;
    int r   = gid - cnn * PER_CNN;
    const float* w0 = cnn ? w0b : w0a;
    const float* w1 = cnn ? w1b : w1a;
    const float* w2 = cnn ? w2b : w2a;
    __bf16* out = wfrag + (size_t)cnn * PER_CNN;

    float val = 0.0f;
    if (r < 2048) {                       // conv0: fused (ci,k) K-dim, kk = ci*5+k (<10)
        int fid = r >> 9, el = r & 511;
        int lane = el >> 4, e = el & 15;
        int m = lane & 15;
        int kidx = a_kidx(lane, e);
        int co = fid * 16 + m;
        if (kidx < 10) {
            int ci = kidx / 5, k = kidx % 5;
            val = w0[(co * 2 + ci) * 5 + k];
        }
    } else if (r < 2048 + 40960) {        // conv1: [ct(8)][cc(2)][k(5)]
        int rr = r - 2048;
        int fid = rr >> 9, el = rr & 511;
        int lane = el >> 4, e = el & 15;
        int m = lane & 15;
        int kidx = a_kidx(lane, e);
        int ct = fid / 10, cc = (fid / 5) % 2, k = fid % 5;
        int co = ct * 16 + m, ci = cc * 32 + kidx;
        val = w1[(co * 64 + ci) * 5 + k];
    } else {                              // conv2: [ct(16)][cc(4)][k(5)]
        int rr = r - 43008;
        int fid = rr >> 9, el = rr & 511;
        int lane = el >> 4, e = el & 15;
        int m = lane & 15;
        int kidx = a_kidx(lane, e);
        int ct = fid / 20, cc = (fid / 5) % 4, k = fid % 5;
        int co = ct * 16 + m, ci = cc * 32 + kidx;
        val = w2[(co * 128 + ci) * 5 + k];
    }
    out[r] = (__bf16)val;
}

// ---------------- conv0: [B,2,4096] fp32 -> [B][2048][64] bf16, K fused to 10 ------
__global__ void conv0_kernel(const float* __restrict__ x, const __bf16* __restrict__ afrag,
                             const float* __restrict__ bias, __bf16* __restrict__ h1) {
    int tid = threadIdx.x, lane = tid & 31;
    int gwave = blockIdx.x * (blockDim.x >> 5) + (tid >> 5);
    int ct = gwave & 3;
    int tt = (gwave >> 2) & 127;
    int b  = gwave >> 9;
    int hi = lane >> 4, n = lane & 15;
    int t  = tt * 16 + n;

    v16bf a = *(const v16bf*)(afrag + ((size_t)ct * 512 + lane * 16));

    FragB bf;
    bf.q[0] = make_uint4(0u, 0u, 0u, 0u);
    bf.q[1] = make_uint4(0u, 0u, 0u, 0u);
    if (hi == 0) {   // B lanes 16-31 carry K=16..31 which are zero-padded
        const float* xb = x + (size_t)b * 2 * 4096;
        #pragma unroll
        for (int e = 0; e < 10; ++e) {
            int ci = e / 5, k = e % 5;
            int pos = 2 * t - 1 + k;
            float v = (pos >= 0 && pos < 4096) ? xb[ci * 4096 + pos] : 0.0f;
            bf.e[e] = (__bf16)v;
        }
    }

    v8f c;
    #pragma unroll
    for (int i = 0; i < 8; ++i) c[i] = 0.0f;
    c = wmma_bf16(a, bf.v, c);

    int co0 = ct * 16 + hi * 8;
    const float4* bp = (const float4*)(bias + co0);
    float4 bb0 = bp[0], bb1 = bp[1];
    float bl[8] = {bb0.x, bb0.y, bb0.z, bb0.w, bb1.x, bb1.y, bb1.z, bb1.w};
    Pack8 p;
    #pragma unroll
    for (int r = 0; r < 8; ++r) {
        float v = c[r] + bl[r];
        p.e[r] = (__bf16)(v > 0.0f ? v : 0.0f);
    }
    *(uint4*)(h1 + (((size_t)b * 2048 + t) * 64 + co0)) = p.q;
}

// ---------------- conv1: h1[B][2048][64] -> h2[B][1024][128], TN=4 t-tiles/wave ----
__global__ void conv1_kernel(const __bf16* __restrict__ h1, const __bf16* __restrict__ afrag,
                             const float* __restrict__ bias, __bf16* __restrict__ h2) {
    int tid = threadIdx.x, lane = tid & 31;
    int gwave = blockIdx.x * (blockDim.x >> 5) + (tid >> 5);
    int ct  = gwave & 7;
    int ttg = (gwave >> 3) & 15;
    int b   = gwave >> 7;
    int hi = lane >> 4, n = lane & 15;

    v8f acc[4];
    #pragma unroll
    for (int j = 0; j < 4; ++j)
        #pragma unroll
        for (int i = 0; i < 8; ++i) acc[j][i] = 0.0f;

    const __bf16* h1b = h1 + (size_t)b * 2048 * 64;
    #pragma unroll
    for (int k = 0; k < 5; ++k) {
        #pragma unroll
        for (int cc = 0; cc < 2; ++cc) {
            v16bf a = *(const v16bf*)(afrag + (((size_t)(ct * 2 + cc) * 5 + k) * 512 + lane * 16));
            #pragma unroll
            for (int j = 0; j < 4; ++j) {
                int t = ttg * 64 + j * 16 + n;
                int pos = 2 * t - 1 + k;
                FragB bf;
                if (pos >= 0 && pos < 2048) {
                    bf.v = *(const v16bf*)(h1b + ((size_t)pos * 64 + cc * 32 + hi * 16));
                } else {
                    bf.q[0] = make_uint4(0u, 0u, 0u, 0u);
                    bf.q[1] = make_uint4(0u, 0u, 0u, 0u);
                }
                acc[j] = wmma_bf16(a, bf.v, acc[j]);
            }
        }
    }

    int co0 = ct * 16 + hi * 8;
    const float4* bp = (const float4*)(bias + co0);
    float4 bb0 = bp[0], bb1 = bp[1];
    float bl[8] = {bb0.x, bb0.y, bb0.z, bb0.w, bb1.x, bb1.y, bb1.z, bb1.w};
    #pragma unroll
    for (int j = 0; j < 4; ++j) {
        int t = ttg * 64 + j * 16 + n;
        Pack8 p;
        #pragma unroll
        for (int r = 0; r < 8; ++r) {
            float v = acc[j][r] + bl[r];
            p.e[r] = (__bf16)(v > 0.0f ? v : 0.0f);
        }
        *(uint4*)(h2 + (((size_t)b * 1024 + t) * 128 + co0)) = p.q;
    }
}

// ---------------- conv2 + fused bias/ReLU/mean-pool: h2 -> pooled[B][256] (sum) ----
// one wave owns (b, 16-channel strip): deterministic, no atomics
__global__ void conv2pool_kernel(const __bf16* __restrict__ h2, const __bf16* __restrict__ afrag,
                                 const float* __restrict__ bias, float* __restrict__ pooled) {
    int tid = threadIdx.x, lane = tid & 31;
    int gwave = blockIdx.x * (blockDim.x >> 5) + (tid >> 5);
    int ct = gwave & 15;
    int b  = gwave >> 4;
    int hi = lane >> 4, n = lane & 15;
    int co0 = ct * 16 + hi * 8;

    const float4* bp = (const float4*)(bias + co0);
    float4 bb0 = bp[0], bb1 = bp[1];
    float bl[8] = {bb0.x, bb0.y, bb0.z, bb0.w, bb1.x, bb1.y, bb1.z, bb1.w};
    float s[8];
    #pragma unroll
    for (int r = 0; r < 8; ++r) s[r] = 0.0f;

    const __bf16* h2b = h2 + (size_t)b * 1024 * 128;
    for (int tg = 0; tg < 8; ++tg) {           // 8 groups x 4 tiles x 16 = 512 positions
        v8f acc[4];
        #pragma unroll
        for (int j = 0; j < 4; ++j)
            #pragma unroll
            for (int i = 0; i < 8; ++i) acc[j][i] = 0.0f;
        #pragma unroll
        for (int k = 0; k < 5; ++k) {
            #pragma unroll
            for (int cc = 0; cc < 4; ++cc) {
                v16bf a = *(const v16bf*)(afrag + (((size_t)(ct * 4 + cc) * 5 + k) * 512 + lane * 16));
                #pragma unroll
                for (int j = 0; j < 4; ++j) {
                    int t = tg * 64 + j * 16 + n;
                    int pos = 2 * t - 1 + k;
                    FragB bf;
                    if (pos >= 0 && pos < 1024) {
                        bf.v = *(const v16bf*)(h2b + ((size_t)pos * 128 + cc * 32 + hi * 16));
                    } else {
                        bf.q[0] = make_uint4(0u, 0u, 0u, 0u);
                        bf.q[1] = make_uint4(0u, 0u, 0u, 0u);
                    }
                    acc[j] = wmma_bf16(a, bf.v, acc[j]);
                }
            }
        }
        #pragma unroll
        for (int j = 0; j < 4; ++j)
            #pragma unroll
            for (int r = 0; r < 8; ++r) {
                float v = acc[j][r] + bl[r];
                s[r] += (v > 0.0f ? v : 0.0f);
            }
    }
    // reduce over the 16 positions held by this lane-half (stays within half: xor<16)
    #pragma unroll
    for (int r = 0; r < 8; ++r) {
        float v = s[r];
        v += __shfl_xor(v, 1);
        v += __shfl_xor(v, 2);
        v += __shfl_xor(v, 4);
        v += __shfl_xor(v, 8);
        s[r] = v;
    }
    if (n == 0) {
        #pragma unroll
        for (int r = 0; r < 8; ++r) pooled[(size_t)b * 256 + co0 + r] = s[r];
    }
}

// ---------------- heads + softmax + routing -> out[B][4] -----------------------
__global__ void head_kernel(const float* __restrict__ pooled1, const float* __restrict__ pooled2,
                            const float* __restrict__ wh1, const float* __restrict__ bh1,
                            const float* __restrict__ wh2, const float* __restrict__ bh2,
                            float* __restrict__ out) {
    int tid = threadIdx.x, lane = tid & 31;
    int b = blockIdx.x * (blockDim.x >> 5) + (tid >> 5);
    if (b >= 512) return;

    float a10 = 0.f, a11 = 0.f, a12 = 0.f, a20 = 0.f, a21 = 0.f;
    const float inv = 1.0f / 512.0f;   // global mean over L=512
    #pragma unroll
    for (int i = 0; i < 8; ++i) {
        int c = lane + i * 32;
        float p1 = pooled1[(size_t)b * 256 + c] * inv;
        float p2 = pooled2[(size_t)b * 256 + c] * inv;
        a10 += p1 * wh1[0 * 256 + c];
        a11 += p1 * wh1[1 * 256 + c];
        a12 += p1 * wh1[2 * 256 + c];
        a20 += p2 * wh2[0 * 256 + c];
        a21 += p2 * wh2[1 * 256 + c];
    }
    #pragma unroll
    for (int m = 16; m >= 1; m >>= 1) {
        a10 += __shfl_xor(a10, m);
        a11 += __shfl_xor(a11, m);
        a12 += __shfl_xor(a12, m);
        a20 += __shfl_xor(a20, m);
        a21 += __shfl_xor(a21, m);
    }
    if (lane == 0) {
        float l0 = a10 + bh1[0], l1 = a11 + bh1[1], l2 = a12 + bh1[2];
        float mx = fmaxf(l0, fmaxf(l1, l2));
        float e0 = expf(l0 - mx), e1 = expf(l1 - mx), e2 = expf(l2 - mx);
        float si = 1.0f / (e0 + e1 + e2);
        float p0 = e0 * si, p1 = e1 * si, p2 = e2 * si;
        int pred = 0; float bm = p0;
        if (p1 > bm) { pred = 1; bm = p1; }
        if (p2 > bm) { pred = 2; }

        float q0 = a20 + bh2[0], q1 = a21 + bh2[1];
        float m2 = fmaxf(q0, q1);
        float f0 = expf(q0 - m2), f1 = expf(q1 - m2);
        float s2 = 1.0f / (f0 + f1);

        float4 o;
        o.x = (pred == 0) ? p0 : 0.0f;
        o.y = (pred == 1) ? p1 : 0.0f;
        o.z = (pred == 2) ? (f0 * s2) : 0.0f;
        o.w = (pred == 2) ? (f1 * s2) : 0.0f;
        *(float4*)(out + (size_t)b * 4) = o;
    }
}

extern "C" void kernel_launch(void* const* d_in, const int* in_sizes, int n_in,
                              void* d_out, int out_size, void* d_ws, size_t ws_size,
                              hipStream_t stream) {
    const float* x    = (const float*)d_in[0];
    const float* l1w0 = (const float*)d_in[1];
    const float* l1b0 = (const float*)d_in[2];
    const float* l1w1 = (const float*)d_in[3];
    const float* l1b1 = (const float*)d_in[4];
    const float* l1w2 = (const float*)d_in[5];
    const float* l1b2 = (const float*)d_in[6];
    const float* l1wh = (const float*)d_in[7];
    const float* l1bh = (const float*)d_in[8];
    const float* l2w0 = (const float*)d_in[9];
    const float* l2b0 = (const float*)d_in[10];
    const float* l2w1 = (const float*)d_in[11];
    const float* l2b1 = (const float*)d_in[12];
    const float* l2w2 = (const float*)d_in[13];
    const float* l2b2 = (const float*)d_in[14];
    const float* l2wh = (const float*)d_in[15];
    const float* l2bh = (const float*)d_in[16];

    char* ws = (char*)d_ws;
    __bf16* h1      = (__bf16*)(ws + 0);                 // [512][2048][64] bf16 = 128 MiB
    __bf16* h2      = (__bf16*)(ws + 134217728ull);      // [512][1024][128] bf16 = 128 MiB
    float*  pooled1 = (float*)(ws + 268435456ull);       // [512][256] f32
    float*  pooled2 = (float*)(ws + 268959744ull);       // [512][256] f32
    __bf16* wfrag   = (__bf16*)(ws + 269484032ull);      // 2 * 206848 bf16 A-fragments

    prep_weights<<<(2 * PER_CNN + 255) / 256, 256, 0, stream>>>(
        l1w0, l1w1, l1w2, l2w0, l2w1, l2w2, wfrag);

    for (int cnn = 0; cnn < 2; ++cnn) {
        const __bf16* wf = wfrag + (size_t)cnn * PER_CNN;
        const float* b0 = cnn ? l2b0 : l1b0;
        const float* b1 = cnn ? l2b1 : l1b1;
        const float* b2 = cnn ? l2b2 : l1b2;
        float* pooled = cnn ? pooled2 : pooled1;
        conv0_kernel<<<32768, 256, 0, stream>>>(x, wf + 0,     b0, h1);
        conv1_kernel<<<8192,  256, 0, stream>>>(h1, wf + 2048,  b1, h2);
        conv2pool_kernel<<<1024, 256, 0, stream>>>(h2, wf + 43008, b2, pooled);
    }

    head_kernel<<<64, 256, 0, stream>>>(pooled1, pooled2, l1wh, l1bh, l2wh, l2bh,
                                        (float*)d_out);
}
